// LSTMModel_68281390072054
// MI455X (gfx1250) — matrix operand order repaired and checked by reading, not compiled, not run on hardware
//
#include <hip/hip_runtime.h>

// ---------------------------------------------------------------------------
// 4-layer LSTM (H=50, B=1024, T=512) + FC head for MI455X (gfx1250).
//
//  * One kernel per layer; each 128-thread WG owns 16 batch rows and runs all
//    512 timesteps internally (latency-bound recurrence -> minimize per-step
//    critical path: 2 barriers, WMMA, and a lean elementwise phase).
//  * gates(16x200) = [h | x_t](16x128 f16) @ Wcat^T(128x200 f16) + bias via
//    v_wmma_f32_16x16x32_f16 (13 N-tiles x 4 k-fragments, f32 accumulate).
//  * Weight (B) fragments preloaded into VGPRs (<=16 v16h per wave); the LDS
//    weight-staging region is reused as the A-staging/gates arena.
//  * Cell state lives in per-thread registers (fixed (m,j) ownership).
//  * Activations use the CDNA5 TANH transcendental (no IEEE divides).
//  * Inter-layer handoff: in-place f16 h-sequence [T][B][64] in d_ws (64 MiB).
// ---------------------------------------------------------------------------

typedef _Float16 half_t;
typedef __attribute__((ext_vector_type(16))) _Float16 v16h;
typedef __attribute__((ext_vector_type(8)))  float    v8f;

constexpr int kB   = 1024;  // batch
constexpr int kT   = 512;   // timesteps
constexpr int kH   = 50;    // hidden
constexpr int kG   = 200;   // 4*H gates
constexpr int kKP  = 128;   // padded K = 64(h) + 64(x)
constexpr int kNT  = 13;    // ceil(200/16) N-tiles
constexpr int kBT  = 16;    // batch tile per workgroup
constexpr int kThr = 128;   // 4 waves
constexpr int kGS  = 208;   // gate LDS row stride (floats)
constexpr int kHS  = 64;    // h-sequence row stride (halves)

__device__ __forceinline__ float fast_tanh(float x) {
#if __has_builtin(__builtin_amdgcn_tanhf)
    return __builtin_amdgcn_tanhf(x);          // v_tanh_f32 (CDNA5 TRANS op)
#else
    const float e = __expf(2.f * x);
    return 1.f - 2.f * __builtin_amdgcn_rcpf(e + 1.f);
#endif
}
__device__ __forceinline__ float fast_sigmoid(float x) {
    return 0.5f * fast_tanh(0.5f * x) + 0.5f;  // no division
}

__global__ __launch_bounds__(kThr, 1)
void lstm_layer_kernel(const float* __restrict__ Wih, const float* __restrict__ Whh,
                       const float* __restrict__ bih, const float* __restrict__ bhh,
                       const float* __restrict__ x0,  // non-null only for layer 0
                       half_t* __restrict__ hseq,     // [T][B][64] f16, in-place
                       int in_sz)
{
    // Phase-1 region (weight staging) reused as the phase-2 arena.
    __shared__ __align__(16) char smem[52032];
    half_t* WstH  = (half_t*)smem;                       // [200][128] f16
    float*  biasL = (float*)(smem + 200 * kKP * 2);      // [208] f32
    half_t* Ast   = (half_t*)smem;                       // [16][128] f16: h|x
    float*  Gst   = (float*)(smem + kBT * kKP * 2);      // [16][208] f32 gates

    const int tid     = threadIdx.x;
    const int lane    = tid & 31;
    const int wave    = __builtin_amdgcn_readfirstlane(tid >> 5);  // uniform
    const int halfsel = lane >> 4;   // 0: lanes 0-15, 1: lanes 16-31
    const int lrow    = lane & 15;
    const int bb      = blockIdx.x * kBT;
    const int mrow    = tid >> 3;    // elementwise: fixed row, 8 threads/row
    const int jbase   = tid & 7;     // elementwise: j = jbase + 8k

    // ---- Phase 1: stage Wcat = [Whh | Wih] (f32->f16) + bias into LDS ----
    for (int i = tid; i < kG * kKP; i += kThr) WstH[i] = (half_t)0.f;
    __syncthreads();
    for (int i = tid; i < kG * kH; i += kThr) {
        int n = i / kH, k = i % kH;
        WstH[n * kKP + k] = (half_t)Whh[n * kH + k];
    }
    for (int i = tid; i < kG * in_sz; i += kThr) {
        int n = i / in_sz, k = i % in_sz;
        WstH[n * kKP + 64 + k] = (half_t)Wih[n * in_sz + k];
    }
    for (int i = tid; i < 208; i += kThr)
        biasL[i] = (i < kG) ? (bih[i] + bhh[i]) : 0.f;
    __syncthreads();

    // ---- Load B fragments into VGPRs (fully unrolled, constant indices) ----
    // B layout (16-bit, 32x16 KxN): lane column N = nt*16 + lrow; halves 0..7
    // hold K = kb*32 + 8*halfsel + (0..7), halves 8..15 = +16 (matches A).
    v16h  bfrag[4][4];
    float biasr[4];
    #pragma unroll
    for (int i = 0; i < 4; ++i) {
        const int nt = wave + 4 * i;            // scalar -> scalar branch
        if (nt < kNT) {
            const int n = nt * 16 + lrow;
            biasr[i] = biasL[n];
            #pragma unroll
            for (int kb = 0; kb < 4; ++kb) {
                const half_t* p = &WstH[n * kKP + kb * 32 + 8 * halfsel];
                #pragma unroll
                for (int j = 0; j < 16; ++j) bfrag[i][kb][j] = p[j < 8 ? j : j + 8];
            }
        }
    }
    __syncthreads();

    // ---- Phase 2 init: h|x staging zero (covers all pads) ----
    for (int i = tid; i < kBT * kKP; i += kThr) Ast[i] = (half_t)0.f;
    __syncthreads();

    // Cell state in registers: thread owns (mrow, jbase+8k), k=0..6.
    float creg[7];
    #pragma unroll
    for (int k = 0; k < 7; ++k) creg[k] = 0.f;

    // ---- Time loop (sequential recurrence) ----
    for (int t = 0; t < kT; ++t) {
        // Stage x_t into A cols [64, 64+in_sz).
        if (x0) {  // layer 0: x is (B,T,7) f32; 16 rows x 7 cols, no div
            if (jbase < 7) {
                Ast[mrow * kKP + 64 + jbase] =
                    (half_t)x0[((size_t)(bb + mrow) * kT + t) * 7 + jbase];
            }
        } else {   // layers 1-3: prev layer's h (f16, in place), 2 halves/load
            const uint32_t* hp = (const uint32_t*)
                (hseq + ((size_t)t * kB + bb + mrow) * kHS);
            uint32_t* ap = (uint32_t*)&Ast[mrow * kKP + 64];
            #pragma unroll
            for (int k = 0; k < 4; ++k) {
                const int p = jbase + 8 * k;    // half-pair index, 25 per row
                if (p < 25) ap[p] = hp[p];
            }
        }
        __syncthreads();  // x staged; h from previous step fully written

        // A fragments. Layout (16-bit, 16x32 MxK): row M = lrow for both lane
        // halves; halves 0..7 = K kb*32+8*halfsel+(0..7), 8..15 = +16.
        v16h afrag[4];
        #pragma unroll
        for (int kb = 0; kb < 4; ++kb) {
            const half_t* p1 = &Ast[lrow * kKP + kb * 32 + 8 * halfsel];
            const half_t* p2 = p1 + 16;
            #pragma unroll
            for (int j = 0; j < 8; ++j) {
                afrag[kb][j]     = p1[j];
                afrag[kb][8 + j] = p2[j];
            }
        }

        // WMMA: gates = [h|x] @ Wcat^T + bias (C seeded with bias).
        #pragma unroll
        for (int i = 0; i < 4; ++i) {
            const int nt = wave + 4 * i;        // uniform guard, EXEC all-1s
            if (nt < kNT) {
                v8f acc;
                #pragma unroll
                for (int v = 0; v < 8; ++v) acc[v] = biasr[i];
                #pragma unroll
                for (int kb = 0; kb < 4; ++kb)
                    acc = __builtin_amdgcn_wmma_f32_16x16x32_f16(
                        false, afrag[kb], false, bfrag[i][kb],
                        (short)0, acc, false, false);
                // C layout: VGPR v, lanes 0-15 -> M=v, lanes 16-31 -> M=8+v.
                #pragma unroll
                for (int v = 0; v < 8; ++v)
                    Gst[(v + 8 * halfsel) * kGS + nt * 16 + lrow] = acc[v];
            }
        }
        __syncthreads();  // gates complete (also fences A-frag LDS reads)

        // Elementwise LSTM cell: fixed (mrow, j) slots, c in registers,
        // TANH-based activations (no divides, no div/mod addressing).
        {
            const float* g = &Gst[mrow * kGS];
            half_t* hrow = hseq + ((size_t)t * kB + bb + mrow) * kHS;
            #pragma unroll
            for (int k = 0; k < 7; ++k) {
                const int j = jbase + 8 * k;
                if (j < kH) {
                    const float ig = fast_sigmoid(g[j]);
                    const float fg = fast_sigmoid(g[kH + j]);
                    const float gg = fast_tanh(g[2 * kH + j]);
                    const float og = fast_sigmoid(g[3 * kH + j]);
                    const float c  = fg * creg[k] + ig * gg;
                    const float h  = og * fast_tanh(c);
                    creg[k] = c;
                    Ast[mrow * kKP + j] = (half_t)h;  // next step's A h-region
                    hrow[j] = (half_t)h;
                }
            }
        }
        // No trailing barrier: next iteration's x-stage touches only disjoint
        // A columns; its barrier orders everything else.
    }
}

// ---- FC head: relu(h @ W1^T + b1) @ W2^T + b2, one thread per batch row ----
__global__ __launch_bounds__(256)
void fc_head_kernel(const half_t* __restrict__ hseq,
                    const float* __restrict__ W1, const float* __restrict__ b1,
                    const float* __restrict__ W2, const float* __restrict__ b2,
                    float* __restrict__ out)
{
    __shared__ float W1s[25 * 50];
    __shared__ float b1s[25];
    __shared__ float W2s[25];
    __shared__ float b2s;
    const int tid = threadIdx.x;
    for (int i = tid; i < 25 * 50; i += 256) W1s[i] = W1[i];
    if (tid < 25) { b1s[tid] = b1[tid]; W2s[tid] = W2[tid]; }
    if (tid == 0) b2s = b2[0];
    __syncthreads();

    const int b = blockIdx.x * 256 + tid;
    float h[50];
    #pragma unroll
    for (int j = 0; j < 50; ++j)
        h[j] = (float)hseq[((size_t)(kT - 1) * kB + b) * kHS + j];

    float o = b2s;
    for (int r = 0; r < 25; ++r) {
        float a = b1s[r];
        #pragma unroll
        for (int j = 0; j < 50; ++j) a += W1s[r * 50 + j] * h[j];
        o += W2s[r] * fmaxf(a, 0.f);
    }
    out[b] = o;
}

extern "C" void kernel_launch(void* const* d_in, const int* in_sizes, int n_in,
                              void* d_out, int out_size, void* d_ws, size_t ws_size,
                              hipStream_t stream)
{
    (void)in_sizes; (void)n_in; (void)out_size; (void)ws_size;
    const float* x = (const float*)d_in[0];
    half_t* hseq = (half_t*)d_ws;  // [T][B][64] f16 = 64 MiB, reused in place

    for (int l = 0; l < 4; ++l) {
        const float* Wih = (const float*)d_in[1 + 4 * l];
        const float* Whh = (const float*)d_in[2 + 4 * l];
        const float* bih = (const float*)d_in[3 + 4 * l];
        const float* bhh = (const float*)d_in[4 + 4 * l];
        lstm_layer_kernel<<<dim3(kB / kBT), dim3(kThr), 0, stream>>>(
            Wih, Whh, bih, bhh, (l == 0) ? x : nullptr, hseq, (l == 0) ? 7 : kH);
    }

    const float* W1 = (const float*)d_in[17];
    const float* b1 = (const float*)d_in[18];
    const float* W2 = (const float*)d_in[19];
    const float* b2 = (const float*)d_in[20];
    fc_head_kernel<<<dim3(kB / 256), dim3(256), 0, stream>>>(
        hseq, W1, b1, W2, b2, (float*)d_out);
}
